// MaskedLinear_11965778887222
// MI455X (gfx1250) — matrix operand order repaired and checked by reading, not compiled, not run on hardware
//
#include <hip/hip_runtime.h>
#include <hip/hip_bf16.h>

typedef __attribute__((ext_vector_type(16))) __bf16 v16bf;
typedef __attribute__((ext_vector_type(8)))  float  v8f;

union FragU { uint4 q[2]; v16bf v; };

// round-to-nearest-even fp32 -> bf16, packed pair
__device__ __forceinline__ unsigned pack_bf16(float a, float b) {
    unsigned ua = __float_as_uint(a), ub = __float_as_uint(b);
    ua += 0x7FFFu + ((ua >> 16) & 1u);
    ub += 0x7FFFu + ((ub >> 16) & 1u);
    return (ua >> 16) | (ub & 0xFFFF0000u);
}
__device__ __forceinline__ unsigned short cvt_bf16(float a) {
    unsigned ua = __float_as_uint(a);
    ua += 0x7FFFu + ((ua >> 16) & 1u);
    return (unsigned short)(ua >> 16);
}

// A fragment 16(M)x32(K) bf16 from LDS (row-major, stride lda elements).
// Lane L<16: row=L, V0..7 = K[0..15]; lanes 16..31: K[16..31].
__device__ __forceinline__ v16bf load_frag_a(const unsigned short* As, int lda,
                                             int mr, int kk, int lane) {
    int row = mr + (lane & 15);
    int kh  = (lane >> 4) * 16;
    const uint4* p = (const uint4*)(As + row * lda + kk + kh);
    FragU f;
    f.q[0] = p[0];   // K 0..7   (relative)
    f.q[1] = p[1];   // K 8..15  (+16 bytes)
    return f.v;
}

// B fragment 32(K)x16(N) bf16 from LDS stored N-major (BT[n][k], stride ldb).
// Lane L<16: col=L, V0..7 = K[0..15]; lanes 16..31: K[16..31].
__device__ __forceinline__ v16bf load_frag_b(const unsigned short* Bs, int ldb,
                                             int nc, int kk, int lane) {
    int col = nc + (lane & 15);
    int kh  = (lane >> 4) * 16;
    const uint4* p = (const uint4*)(Bs + col * ldb + kk + kh);
    FragU f;
    f.q[0] = p[0];
    f.q[1] = p[1];
    return f.v;
}

__device__ __forceinline__ v8f wmma_bf16(v16bf a, v16bf b, v8f c) {
    return __builtin_amdgcn_wmma_f32_16x16x32_bf16(false, a, false, b,
                                                   (short)0, c, false, false);
}

// ---------------------------------------------------------------------------
// GEMM1: h[256,8192] = relu(x[256,4096] @ (W1*mask1) + b1), stored as bf16.
// Column block j (128 wide) only uses K < 64*(j+1)  (block-causal mask).
// Tile: 128(M) x 128(N), 8 waves, each wave 32x64 (2x4 subtiles).
// ---------------------------------------------------------------------------
__global__ __launch_bounds__(256) void gemm1_kernel(
    const float* __restrict__ x, const float* __restrict__ W1,
    const float* __restrict__ b1, unsigned short* __restrict__ h) {
    __shared__ __align__(16) unsigned short As[128 * 72];
    __shared__ __align__(16) unsigned short Bs[128 * 72];

    const int tid  = threadIdx.x;
    const int lane = tid & 31;
    const int wave = tid >> 5;
    const int j    = blockIdx.x;           // hidden seq block, 0..63
    const int n0   = j * 128;
    const int m0   = blockIdx.y * 128;
    const int Kmax = 64 * (j + 1);         // causal mask => K limit
    const int wm   = 32 * (wave & 3);
    const int wn   = 64 * (wave >> 2);

    v8f acc[2][4] = {};

    const int nb = (tid & 31) * 4;         // B staging: col base
    for (int kc = 0; kc < Kmax; kc += 64) {
        // --- stage A tile [128 x 64] fp32 -> bf16 LDS ---
#pragma unroll
        for (int q = 0; q < 8; ++q) {
            int idx = tid + 256 * q;
            int row = idx >> 4, col = (idx & 15) * 4;
            float4 v = *(const float4*)(x + (size_t)(m0 + row) * 4096 + kc + col);
            uint2 pk = make_uint2(pack_bf16(v.x, v.y), pack_bf16(v.z, v.w));
            *(uint2*)(As + row * 72 + col) = pk;
        }
        // --- stage B tile [64 x 128] fp32 -> bf16 LDS, transposed (N-major) ---
#pragma unroll
        for (int q = 0; q < 2; ++q) {
            int kb = (tid >> 5) * 4 + 32 * q;
            const float* wp = W1 + (size_t)(kc + kb) * 8192 + n0 + nb;
            float4 r0 = *(const float4*)(wp);
            float4 r1 = *(const float4*)(wp + 8192);
            float4 r2 = *(const float4*)(wp + 2 * 8192);
            float4 r3 = *(const float4*)(wp + 3 * 8192);
            if (kc + 64 < Kmax)
                __builtin_prefetch(wp + (size_t)64 * 8192, 0, 1);
            *(uint2*)(Bs + (nb + 0) * 72 + kb) =
                make_uint2(pack_bf16(r0.x, r1.x), pack_bf16(r2.x, r3.x));
            *(uint2*)(Bs + (nb + 1) * 72 + kb) =
                make_uint2(pack_bf16(r0.y, r1.y), pack_bf16(r2.y, r3.y));
            *(uint2*)(Bs + (nb + 2) * 72 + kb) =
                make_uint2(pack_bf16(r0.z, r1.z), pack_bf16(r2.z, r3.z));
            *(uint2*)(Bs + (nb + 3) * 72 + kb) =
                make_uint2(pack_bf16(r0.w, r1.w), pack_bf16(r2.w, r3.w));
        }
        __syncthreads();
        // --- compute: 2 K-chunks of 32 ---
#pragma unroll
        for (int kk = 0; kk < 64; kk += 32) {
            v16bf a0 = load_frag_a(As, 72, wm + 0,  kk, lane);
            v16bf a1 = load_frag_a(As, 72, wm + 16, kk, lane);
            v16bf bf0 = load_frag_b(Bs, 72, wn + 0,  kk, lane);
            v16bf bf1 = load_frag_b(Bs, 72, wn + 16, kk, lane);
            v16bf bf2 = load_frag_b(Bs, 72, wn + 32, kk, lane);
            v16bf bf3 = load_frag_b(Bs, 72, wn + 48, kk, lane);
            acc[0][0] = wmma_bf16(a0, bf0, acc[0][0]);
            acc[0][1] = wmma_bf16(a0, bf1, acc[0][1]);
            acc[0][2] = wmma_bf16(a0, bf2, acc[0][2]);
            acc[0][3] = wmma_bf16(a0, bf3, acc[0][3]);
            acc[1][0] = wmma_bf16(a1, bf0, acc[1][0]);
            acc[1][1] = wmma_bf16(a1, bf1, acc[1][1]);
            acc[1][2] = wmma_bf16(a1, bf2, acc[1][2]);
            acc[1][3] = wmma_bf16(a1, bf3, acc[1][3]);
        }
        __syncthreads();
    }

    // epilogue: bias + relu, store bf16 h
#pragma unroll
    for (int s = 0; s < 2; ++s)
#pragma unroll
        for (int t = 0; t < 4; ++t)
#pragma unroll
            for (int r = 0; r < 8; ++r) {
                int row = m0 + wm + 16 * s + r + 8 * (lane >> 4);
                int col = n0 + wn + 16 * t + (lane & 15);
                float v = acc[s][t][r] + b1[col];
                v = v > 0.0f ? v : 0.0f;
                h[(size_t)row * 8192 + col] = cvt_bf16(v);
            }
}

// ---------------------------------------------------------------------------
// GEMM2: out[256,4096] = h_bf16[256,8192] @ (W2*mask2) + b2  (fp32 out).
// Column block j (64 wide) only uses K < 128*(j+1).
// Tile: 128(M) x 64(N), 8 waves, each wave 16x64 (1x4 subtiles).
// A tile is already bf16 in the exact LDS layout -> use the CDNA5 async
// global->LDS DMA path (ASYNCcnt) instead of a VGPR round-trip.
// ---------------------------------------------------------------------------
__global__ __launch_bounds__(256) void gemm2_kernel(
    const unsigned short* __restrict__ h, const float* __restrict__ W2,
    const float* __restrict__ b2, float* __restrict__ out) {
    __shared__ __align__(16) unsigned short As[128 * 72];
    __shared__ __align__(16) unsigned short Bs[64 * 72];

    const int tid  = threadIdx.x;
    const int lane = tid & 31;
    const int wave = tid >> 5;
    const int j    = blockIdx.x;            // output seq block, 0..63
    const int n0   = j * 64;
    const int m0   = blockIdx.y * 128;
    const int Kmax = 128 * (j + 1);
    const int mr   = 16 * wave;

    v8f acc[4] = {};

    const int nb = (tid & 15) * 4;
    const int kb = (tid >> 4) * 4;
    for (int kc = 0; kc < Kmax; kc += 64) {
        // --- stage A tile [128 x 64] bf16 via async global->LDS b128 ---
#pragma unroll
        for (int q = 0; q < 4; ++q) {
            int idx = tid + 256 * q;
            int row = idx >> 3, cq = (idx & 7) * 8;
            unsigned lds_off =
                (unsigned)(unsigned long long)&As[row * 72 + cq];
            const unsigned short* gp =
                h + (size_t)(m0 + row) * 8192 + kc + cq;
            asm volatile("global_load_async_to_lds_b128 %0, %1, off"
                         :: "v"(lds_off), "v"(gp)
                         : "memory");
        }
        // --- stage B tile [64 x 64] fp32 -> bf16 LDS, transposed ---
        {
            const float* wp = W2 + (size_t)(kc + kb) * 4096 + n0 + nb;
            float4 r0 = *(const float4*)(wp);
            float4 r1 = *(const float4*)(wp + 4096);
            float4 r2 = *(const float4*)(wp + 2 * 4096);
            float4 r3 = *(const float4*)(wp + 3 * 4096);
            if (kc + 64 < Kmax)
                __builtin_prefetch(wp + (size_t)64 * 4096, 0, 1);
            *(uint2*)(Bs + (nb + 0) * 72 + kb) =
                make_uint2(pack_bf16(r0.x, r1.x), pack_bf16(r2.x, r3.x));
            *(uint2*)(Bs + (nb + 1) * 72 + kb) =
                make_uint2(pack_bf16(r0.y, r1.y), pack_bf16(r2.y, r3.y));
            *(uint2*)(Bs + (nb + 2) * 72 + kb) =
                make_uint2(pack_bf16(r0.z, r1.z), pack_bf16(r2.z, r3.z));
            *(uint2*)(Bs + (nb + 3) * 72 + kb) =
                make_uint2(pack_bf16(r0.w, r1.w), pack_bf16(r2.w, r3.w));
        }
        // async A copies must land in LDS before anyone reads the tile
        asm volatile("s_wait_asynccnt 0x0" ::: "memory");
        __syncthreads();
#pragma unroll
        for (int kk = 0; kk < 64; kk += 32) {
            v16bf a0  = load_frag_a(As, 72, mr, kk, lane);
            v16bf bf0 = load_frag_b(Bs, 72, 0,  kk, lane);
            v16bf bf1 = load_frag_b(Bs, 72, 16, kk, lane);
            v16bf bf2 = load_frag_b(Bs, 72, 32, kk, lane);
            v16bf bf3 = load_frag_b(Bs, 72, 48, kk, lane);
            acc[0] = wmma_bf16(a0, bf0, acc[0]);
            acc[1] = wmma_bf16(a0, bf1, acc[1]);
            acc[2] = wmma_bf16(a0, bf2, acc[2]);
            acc[3] = wmma_bf16(a0, bf3, acc[3]);
        }
        __syncthreads();
    }

#pragma unroll
    for (int t = 0; t < 4; ++t)
#pragma unroll
        for (int r = 0; r < 8; ++r) {
            int row = m0 + mr + r + 8 * (lane >> 4);
            int col = n0 + 16 * t + (lane & 15);
            out[(size_t)row * 4096 + col] = acc[t][r] + b2[col];
        }
}

extern "C" void kernel_launch(void* const* d_in, const int* in_sizes, int n_in,
                              void* d_out, int out_size, void* d_ws, size_t ws_size,
                              hipStream_t stream) {
    const float* x  = (const float*)d_in[0];   // [256, 64, 64]
    const float* W1 = (const float*)d_in[1];   // [4096, 8192]
    const float* b1 = (const float*)d_in[2];   // [8192]
    const float* W2 = (const float*)d_in[3];   // [8192, 4096]
    const float* b2 = (const float*)d_in[4];   // [4096]
    unsigned short* h = (unsigned short*)d_ws; // bf16 [256, 8192] = 4 MB
    float* out = (float*)d_out;                // [256, 64, 64] fp32

    gemm1_kernel<<<dim3(64, 2), 256, 0, stream>>>(x, W1, b1, h);
    gemm2_kernel<<<dim3(64, 2), 256, 0, stream>>>(h, W2, b2, out);
}
